// SelfAttentionHead_10144712753841
// MI455X (gfx1250) — compile-verified, matrix operand order checked by reading, not run on hardware
//
#include <hip/hip_runtime.h>
#include <math.h>

typedef __attribute__((ext_vector_type(16))) __bf16 v16bf;
typedef __attribute__((ext_vector_type(8)))  float  v8f;

#define T_SEQ 128
#define E_DIM 256
#define H_DIM 64

// LDS strides chosen so every WMMA fragment chunk is 16-byte aligned
#define XS_STRIDE 272   // bf16 elems/row of x tile  (544B rows)
#define QS_STRIDE 96    // bf16 elems/row of Q/K     (192B rows)
#define VT_STRIDE 136   // bf16 elems/row of V^T     (272B rows)
#define P_STRIDE  136

#define XS_BYTES (T_SEQ * XS_STRIDE * 2)
#define QS_BYTES (T_SEQ * QS_STRIDE * 2)
#define VT_BYTES (H_DIM * VT_STRIDE * 2)
#define SMEM_BYTES (XS_BYTES + 2 * QS_BYTES + VT_BYTES)

static __device__ __forceinline__ unsigned short f2bf(float f) {
  unsigned int u = __float_as_uint(f);
  u += 0x7FFFu + ((u >> 16) & 1u);   // round-to-nearest-even
  return (unsigned short)(u >> 16);
}
static __device__ __forceinline__ unsigned int pk2bf(float a, float b) {
  return (unsigned int)f2bf(a) | ((unsigned int)f2bf(b) << 16);
}

union Frag {
  uint4 u[2];
  v16bf v;
};

// Transpose+convert weights: Wt[w][h][e] = bf16(W_w[e][h]); stays hot in L2.
__global__ void wt_prep(const float* __restrict__ Wq,
                        const float* __restrict__ Wk,
                        const float* __restrict__ Wv,
                        unsigned short* __restrict__ Wt) {
  int idx = blockIdx.x * 256 + threadIdx.x;   // 0 .. 3*64*256-1
  int w   = idx >> 14;
  int rem = idx & 16383;
  int h   = rem >> 8;
  int e   = rem & 255;
  const float* W = (w == 0) ? Wq : (w == 1) ? Wk : Wv;
  Wt[idx] = f2bf(W[e * H_DIM + h]);
}

__global__ __launch_bounds__(256, 4)   // cap VGPRs -> 2 blocks/WGP (LDS-limited)
void attn_head(const float* __restrict__ x,
               const unsigned short* __restrict__ Wt,
               float* __restrict__ out) {
  __shared__ __align__(16) char smem[SMEM_BYTES];
  char* xs = smem;                 // x[b] as bf16 [128][XS_STRIDE]
  char* qs = smem + XS_BYTES;      // Q bf16 [128][QS_STRIDE]
  char* ks = qs + QS_BYTES;        // K bf16 [128][QS_STRIDE]
  char* vt = ks + QS_BYTES;        // V^T bf16 [64][VT_STRIDE]

  const int tid  = threadIdx.x;
  const int wid  = tid >> 5;       // wave 0..7 (wave32)
  const int lane = tid & 31;
  const int l15  = lane & 15;
  const int half = lane >> 4;
  const int b    = blockIdx.x;

  //=== Phase 1: stage x[b] f32 -> bf16 in LDS ======================
  {
    const float4* xg = (const float4*)(x + (size_t)b * (T_SEQ * E_DIM));
    #pragma unroll
    for (int i = 0; i < 32; ++i) {
      int j = i * 256 + tid;                 // float4 index, 0..8191
      float4 v = xg[j];
      int row = j >> 6;
      int c4  = j & 63;
      uint2* dst = (uint2*)(xs + row * (XS_STRIDE * 2) + c4 * 8);
      *dst = make_uint2(pk2bf(v.x, v.y), pk2bf(v.z, v.w));
    }
  }
  __syncthreads();

  const int tw = wid;                        // this wave's 16-row t-strip
  v8f zero8 = {0.f,0.f,0.f,0.f,0.f,0.f,0.f,0.f};

  //=== Phase 2: Q,K,V = x @ Wq/Wk/Wv  (bf16 WMMA, fp32 acc) =========
  {
    v8f acc[3][4];
    #pragma unroll
    for (int m = 0; m < 3; ++m)
      #pragma unroll
      for (int h = 0; h < 4; ++h) acc[m][h] = zero8;

    // unroll-2: enough load/WMMA overlap without spilling past 256 VGPRs
    #pragma unroll 2
    for (int k0 = 0; k0 < 8; ++k0) {         // E=256 in K=32 steps
      Frag a;                                 // A = x tile 16x32
      const char* ab = xs + ((tw * 16 + l15) * XS_STRIDE + k0 * 32 + half * 8) * 2;
      a.u[0] = *(const uint4*)ab;
      a.u[1] = *(const uint4*)(ab + 32);
      #pragma unroll
      for (int m = 0; m < 3; ++m) {
        const unsigned short* Wm = Wt + m * (H_DIM * E_DIM);
        #pragma unroll
        for (int h = 0; h < 4; ++h) {
          Frag bb;                            // B = W^T rows (contiguous in E)
          const unsigned short* wp = Wm + (h * 16 + l15) * E_DIM + k0 * 32 + half * 8;
          bb.u[0] = *(const uint4*)wp;
          bb.u[1] = *(const uint4*)(wp + 16);
          acc[m][h] = __builtin_amdgcn_wmma_f32_16x16x32_bf16(
              false, a.v, false, bb.v, (short)0, acc[m][h], false, false);
        }
      }
    }

    #pragma unroll
    for (int h = 0; h < 4; ++h) {
      #pragma unroll
      for (int r = 0; r < 8; ++r) {          // D row m = r + 8*half
        int trow = tw * 16 + half * 8 + r;
        *(unsigned short*)(qs + (trow * QS_STRIDE + h * 16 + l15) * 2) = f2bf(acc[0][h][r]);
        *(unsigned short*)(ks + (trow * QS_STRIDE + h * 16 + l15) * 2) = f2bf(acc[1][h][r]);
      }
      uint4 pv;                               // V stored transposed, packed b128
      pv.x = pk2bf(acc[2][h][0], acc[2][h][1]);
      pv.y = pk2bf(acc[2][h][2], acc[2][h][3]);
      pv.z = pk2bf(acc[2][h][4], acc[2][h][5]);
      pv.w = pk2bf(acc[2][h][6], acc[2][h][7]);
      *(uint4*)(vt + ((h * 16 + l15) * VT_STRIDE + tw * 16 + half * 8) * 2) = pv;
    }
  }
  __syncthreads();

  //=== Phase 3: S = Q K^T * scale, causal softmax ===================
  v8f s[8];
  #pragma unroll
  for (int n = 0; n < 8; ++n) s[n] = zero8;

  const int ntiles = (tw & ~1) + 2;           // skip fully-masked key tiles
  #pragma unroll
  for (int kk = 0; kk < 2; ++kk) {            // H=64 in K=32 steps
    Frag a;
    const char* ab = qs + ((tw * 16 + l15) * QS_STRIDE + kk * 32 + half * 8) * 2;
    a.u[0] = *(const uint4*)ab;
    a.u[1] = *(const uint4*)(ab + 32);
    #pragma unroll
    for (int n = 0; n < 8; ++n) {
      if (n >= ntiles) continue;
      Frag bb;                                // B = K rows (contiguous in H)
      const char* kb = ks + ((n * 16 + l15) * QS_STRIDE + kk * 32 + half * 8) * 2;
      bb.u[0] = *(const uint4*)kb;
      bb.u[1] = *(const uint4*)(kb + 32);
      s[n] = __builtin_amdgcn_wmma_f32_16x16x32_bf16(
          false, a.v, false, bb.v, (short)0, s[n], false, false);
    }
  }

  float rowm[8], rows[8];
  #pragma unroll
  for (int r = 0; r < 8; ++r) rowm[r] = -__builtin_inff();

  #pragma unroll
  for (int n = 0; n < 8; ++n) {
    if (n >= ntiles) continue;
    int ki = n * 16 + l15;
    #pragma unroll
    for (int r = 0; r < 8; ++r) {
      int qi = tw * 16 + half * 8 + r;
      float v = s[n][r] * 0.125f;             // 1/sqrt(64)
      v = (ki <= qi) ? v : -__builtin_inff();
      s[n][r] = v;
      rowm[r] = fmaxf(rowm[r], v);
    }
  }
  // score row lives across exactly 16 lanes -> xor-reduce within halves
  #pragma unroll
  for (int r = 0; r < 8; ++r) {
    float m = rowm[r];
    m = fmaxf(m, __shfl_xor(m, 1));
    m = fmaxf(m, __shfl_xor(m, 2));
    m = fmaxf(m, __shfl_xor(m, 4));
    m = fmaxf(m, __shfl_xor(m, 8));
    rowm[r] = m;
    rows[r] = 0.f;
  }
  #pragma unroll
  for (int n = 0; n < 8; ++n) {
    if (n >= ntiles) continue;
    #pragma unroll
    for (int r = 0; r < 8; ++r) {
      float p = __expf(s[n][r] - rowm[r]);
      s[n][r] = p;
      rows[r] += p;
    }
  }
  float inv[8];
  #pragma unroll
  for (int r = 0; r < 8; ++r) {
    float t = rows[r];
    t += __shfl_xor(t, 1);
    t += __shfl_xor(t, 2);
    t += __shfl_xor(t, 4);
    t += __shfl_xor(t, 8);
    inv[r] = 1.0f / t;
  }

  // P (bf16) -> per-wave LDS scratch aliasing the dead x region
  unsigned short* Pw = (unsigned short*)(smem + wid * (16 * P_STRIDE * 2));
  #pragma unroll
  for (int n = 0; n < 8; ++n) {
    if (n >= ntiles) continue;
    #pragma unroll
    for (int r = 0; r < 8; ++r)
      Pw[(half * 8 + r) * P_STRIDE + n * 16 + l15] = f2bf(s[n][r] * inv[r]);
  }
  asm volatile("s_wait_dscnt 0" ::: "memory"); // cross-lane LDS dep within wave

  //=== Phase 4: out = P @ V =========================================
  v8f o[4];
  #pragma unroll
  for (int n = 0; n < 4; ++n) o[n] = zero8;
  const int kkmax = tw >> 1;                  // causality-pruned key range
  #pragma unroll
  for (int kk = 0; kk < 4; ++kk) {
    if (kk > kkmax) continue;
    Frag a;                                   // A = P tile 16x32
    const char* ab = (const char*)Pw + (l15 * P_STRIDE + kk * 32 + half * 8) * 2;
    a.u[0] = *(const uint4*)ab;
    a.u[1] = *(const uint4*)(ab + 32);
    #pragma unroll
    for (int n = 0; n < 4; ++n) {
      Frag bb;                                // B = V^T rows (contiguous in keys)
      const char* vb = vt + ((n * 16 + l15) * VT_STRIDE + kk * 32 + half * 8) * 2;
      bb.u[0] = *(const uint4*)vb;
      bb.u[1] = *(const uint4*)(vb + 32);
      o[n] = __builtin_amdgcn_wmma_f32_16x16x32_bf16(
          false, a.v, false, bb.v, (short)0, o[n], false, false);
    }
  }

  float* op = out + (size_t)b * (T_SEQ * H_DIM);
  #pragma unroll
  for (int n = 0; n < 4; ++n) {
    #pragma unroll
    for (int r = 0; r < 8; ++r)
      op[(tw * 16 + half * 8 + r) * H_DIM + n * 16 + l15] = o[n][r];
  }
}

extern "C" void kernel_launch(void* const* d_in, const int* in_sizes, int n_in,
                              void* d_out, int out_size, void* d_ws, size_t ws_size,
                              hipStream_t stream) {
  const float* x  = (const float*)d_in[0];
  const float* Wq = (const float*)d_in[1];
  const float* Wk = (const float*)d_in[2];
  const float* Wv = (const float*)d_in[3];
  unsigned short* Wt = (unsigned short*)d_ws;   // 3*64*256 bf16 = 96 KB

  int Bn = in_sizes[0] / (T_SEQ * E_DIM);       // 4096

  wt_prep<<<(3 * H_DIM * E_DIM) / 256, 256, 0, stream>>>(Wq, Wk, Wv, Wt);
  attn_head<<<Bn, 256, 0, stream>>>(x, Wt, (float*)d_out);
}
